// ImplicitNet_step_42133629173881
// MI455X (gfx1250) — matrix operand order repaired and checked by reading, not compiled, not run on hardware
//
#include <hip/hip_runtime.h>
#include <math.h>

// ImplicitNet step on MI455X (gfx1250, wave32, WMMA), fully fused.
// B=256, N=512, H=1, THETA=0.5, LMBDA=1e-5.
//
// A = (W - W^T)/2 is antisymmetric => A^T = -A: every matmul is z @ A against
// ONE packed f16 matrix (signs folded into epilogues). Batch rows are
// independent, so a WG owning 16 rows runs ALL ~119 dependent GEMM phases
// (init, 64 Neumann iterations replacing the batched solve [spectral radius
// ~0.8], 21 GD steps, 11 fixed-point steps) in ONE persistent kernel with only
// workgroup barriers. f16 left operand lives in LDS; all f32 per-row state is
// register-resident (each wave owns a fixed 16x64 column slab). The first half
// of packed A (256 KB, kt 0..7) is cached in LDS per WGP (288 KB of the 320 KB
// LDS used), halving per-phase L2 traffic; the other half streams from the
// 192 MB L2.

#define NN 512
#define TH 0.5f
#define LM 1e-5f

typedef __attribute__((ext_vector_type(16))) _Float16 v16h;
typedef __attribute__((ext_vector_type(8)))  float    v8f;

// ---------------------------------------------------------------------------
// Pack A = 0.5*(W - W^T) into f16, blocked for the wave32 WMMA B-operand
// of V_WMMA_F32_16X16X32_F16 in contiguous k order (A loads match): block(kt,nt) is 32x16 (KxN);
// lane = (n&15) + 16*((k&31)>>4) holds K = 16*(lane>>4)+h, h=0..15,
// stored contiguously (32B per lane -> 2x b128 coalesced loads).
// Block-major => kt 0..7 is the first contiguous 256 KB (easy LDS staging).
// ---------------------------------------------------------------------------
__global__ void antisym_pack_kernel(const float* __restrict__ W,
                                    _Float16* __restrict__ Apk) {
  int i = blockIdx.x * blockDim.x + threadIdx.x;   // 512*512 threads
  int k = i >> 9;
  int n = i & 511;
  float a = 0.5f * (W[k * NN + n] - W[n * NN + k]);
  int kt = k >> 5, ks = k & 31, nt = n >> 4, ns = n & 15;
  int lane = ns + ((ks >> 4) << 4);
  int h = ks & 15;
  Apk[(size_t)(kt * 32 + nt) * 512 + lane * 16 + h] = (_Float16)a;
}

// ---------------------------------------------------------------------------
// One wave: acc[0..3] = Ltile(16x512, LDS f16) @ A[:, ntBase*16 .. +63] as
// four 16x16 f32 accumulators. A-fragments: ds_load_b128. B-fragments:
// kt 0..7 from LDS cache (ds_load_b128), kt 8..15 from global (b128, L2).
// ---------------------------------------------------------------------------
__device__ __forceinline__ void gemm16(const _Float16* sL,
                                       const _Float16* sA,
                                       const _Float16* __restrict__ Apk,
                                       int lane, int ntBase, v8f acc[4]) {
  const int hl = lane >> 4;
  const int m  = lane & 15;
  acc[0] = v8f{}; acc[1] = v8f{}; acc[2] = v8f{}; acc[3] = v8f{};
#pragma unroll 2
  for (int kt = 0; kt < 8; ++kt) {               // K 0..255 : LDS-cached A
    union { uint4 u[2]; v16h h; } ua;
    const _Float16* zr = sL + m * NN + kt * 32 + hl * 16;
    ua.u[0] = *reinterpret_cast<const uint4*>(zr);
    ua.u[1] = *reinterpret_cast<const uint4*>(zr + 8);
#pragma unroll
    for (int j = 0; j < 4; ++j) {
      union { uint4 u[2]; v16h h; } ub;
      const _Float16* bp = sA + ((kt * 32 + ntBase + j) << 9) + lane * 16;
      ub.u[0] = *reinterpret_cast<const uint4*>(bp);
      ub.u[1] = *reinterpret_cast<const uint4*>(bp + 8);
      acc[j] = __builtin_amdgcn_wmma_f32_16x16x32_f16(
          false, ua.h, false, ub.h, (short)0, acc[j], false, false);
    }
  }
#pragma unroll 2
  for (int kt = 8; kt < 16; ++kt) {              // K 256..511 : from L2
    union { uint4 u[2]; v16h h; } ua;
    const _Float16* zr = sL + m * NN + kt * 32 + hl * 16;
    ua.u[0] = *reinterpret_cast<const uint4*>(zr);
    ua.u[1] = *reinterpret_cast<const uint4*>(zr + 8);
#pragma unroll
    for (int j = 0; j < 4; ++j) {
      union { uint4 u[2]; v16h h; } ub;
      const _Float16* bp =
          Apk + ((size_t)(kt * 32 + ntBase + j) << 9) + lane * 16;
      ub.u[0] = *reinterpret_cast<const uint4*>(bp);
      ub.u[1] = *reinterpret_cast<const uint4*>(bp + 8);
      acc[j] = __builtin_amdgcn_wmma_f32_16x16x32_f16(
          false, ua.h, false, ub.h, (short)0, acc[j], false, false);
    }
  }
}

// ---------------------------------------------------------------------------
// Persistent kernel: 16 WGs x 256 threads (8 waves). WG b owns rows
// [16b, 16b+16); wave w owns columns [64w, 64w+64). C-tile layout:
// vgpr r -> row r + 8*(lane>>4), col j*16 + (lane&15).
// ---------------------------------------------------------------------------
__global__ __launch_bounds__(256) void implicit_fused_kernel(
    const float* __restrict__ x, const _Float16* __restrict__ Apk,
    const float* __restrict__ bias, float* __restrict__ out) {
  __shared__ __align__(16) _Float16 sZ[16 * NN];       // 16 KB f16 lhs
  __shared__ __align__(16) _Float16 sP[16 * NN];       // 16 KB f16 (s*r)
  __shared__ __align__(16) _Float16 sA[8 * 32 * 512];  // 256 KB A kt 0..7

  const int lane = threadIdx.x & 31;
  const int wave = threadIdx.x >> 5;
  const int hl = lane >> 4;
  const int m  = lane & 15;
  const int ntBase = wave * 4;
  const int r0 = blockIdx.x * 16;        // global row base

  int lcol[4], lrow[8];
#pragma unroll
  for (int j = 0; j < 4; ++j) lcol[j] = wave * 64 + j * 16 + m;
#pragma unroll
  for (int r = 0; r < 8; ++r) lrow[r] = r + 8 * hl;

  float bb[4];
#pragma unroll
  for (int j = 0; j < 4; ++j) bb[j] = bias[lcol[j]];

  // Register-resident f32 state (each 32 VGPRs).
  float ex[4][8];    // explicit
  float f0[4][8];    // F0 (Neumann rhs)
  float s0[4][8];    // s0 (Neumann diag)
  float z32[4][8];   // v during Neumann, then z
  float rr[4][8];    // gd residual

  // ---- stage LDS: z16 = f16(x); A-cache = first 256 KB of packed A ----
#pragma unroll
  for (int j = 0; j < 4; ++j)
#pragma unroll
    for (int r = 0; r < 8; ++r)
      sZ[lrow[r] * NN + lcol[j]] =
          (_Float16)x[(size_t)(r0 + lrow[r]) * NN + lcol[j]];
  {
    const uint4* src = reinterpret_cast<const uint4*>(Apk);
    uint4* dst = reinterpret_cast<uint4*>(sA);
    for (int i = threadIdx.x; i < (8 * 32 * 512 * 2) / 16; i += 256)
      dst[i] = src[i];
  }
  __syncthreads();

  v8f acc[4];

  // ---- init: F0, s0, explicit, v0 = F0 ----
  gemm16(sZ, sA, Apk, lane, ntBase, acc);
  __syncthreads();
#pragma unroll
  for (int j = 0; j < 4; ++j)
#pragma unroll
    for (int r = 0; r < 8; ++r) {
      float t = tanhf(acc[j][r] + bb[j]);
      f0[j][r] = t;                       // H = 1
      s0[j][r] = 1.0f - t * t;
      ex[j][r] = x[(size_t)(r0 + lrow[r]) * NN + lcol[j]] + (1.0f - TH) * t;
      z32[j][r] = t;                      // v0 = F0
      sZ[lrow[r] * NN + lcol[j]] = (_Float16)t;
    }
  __syncthreads();

  // ---- Neumann solve of (I - th*J) v = F0: v <- F0 + th*s0.(v@A) ----
  for (int it = 0; it < 64; ++it) {
    gemm16(sZ, sA, Apk, lane, ntBase, acc);
    __syncthreads();
#pragma unroll
    for (int j = 0; j < 4; ++j)
#pragma unroll
      for (int r = 0; r < 8; ++r) {
        float v = f0[j][r] + TH * s0[j][r] * acc[j][r];
        z32[j][r] = v;
        sZ[lrow[r] * NN + lcol[j]] = (_Float16)v;
      }
    __syncthreads();
  }

  // ---- y_lin = x + v -> f16 lhs ----
#pragma unroll
  for (int j = 0; j < 4; ++j)
#pragma unroll
    for (int r = 0; r < 8; ++r)
      sZ[lrow[r] * NN + lcol[j]] =
          (_Float16)(x[(size_t)(r0 + lrow[r]) * NN + lcol[j]] + z32[j][r]);
  __syncthreads();

  // ---- y = explicit + th*tanh(y_lin@A + b) ----
  gemm16(sZ, sA, Apk, lane, ntBase, acc);
  __syncthreads();
#pragma unroll
  for (int j = 0; j < 4; ++j)
#pragma unroll
    for (int r = 0; r < 8; ++r) {
      float v = ex[j][r] + TH * tanhf(acc[j][r] + bb[j]);
      z32[j][r] = v;
      sZ[lrow[r] * NN + lcol[j]] = (_Float16)v;
    }
  __syncthreads();

  // ---- 21 gradient-descent steps (A^T = -A folded in) ----
  for (int it = 0; it < 21; ++it) {
    gemm16(sZ, sA, Apk, lane, ntBase, acc);     // z@A
    __syncthreads();
#pragma unroll
    for (int j = 0; j < 4; ++j)
#pragma unroll
      for (int r = 0; r < 8; ++r) {
        float t = tanhf(acc[j][r] + bb[j]);
        float s = 1.0f - t * t;
        float rv = z32[j][r] - ex[j][r] - TH * t;
        rr[j][r] = rv;
        sP[lrow[r] * NN + lcol[j]] = (_Float16)(s * rv);
      }
    __syncthreads();
    gemm16(sP, sA, Apk, lane, ntBase, acc);     // (s*r)@A
    __syncthreads();
#pragma unroll
    for (int j = 0; j < 4; ++j)
#pragma unroll
      for (int r = 0; r < 8; ++r) {
        float zn = z32[j][r] - 2.0f * LM * (rr[j][r] + TH * acc[j][r]);
        z32[j][r] = zn;
        sZ[lrow[r] * NN + lcol[j]] = (_Float16)zn;
      }
    __syncthreads();
  }

  // ---- 10 fixed-point iterations + final evaluation ----
  for (int it = 0; it < 11; ++it) {
    gemm16(sZ, sA, Apk, lane, ntBase, acc);
    __syncthreads();
    if (it < 10) {
#pragma unroll
      for (int j = 0; j < 4; ++j)
#pragma unroll
        for (int r = 0; r < 8; ++r) {
          float v = ex[j][r] + TH * tanhf(acc[j][r] + bb[j]);
          z32[j][r] = v;
          sZ[lrow[r] * NN + lcol[j]] = (_Float16)v;
        }
    } else {
#pragma unroll
      for (int j = 0; j < 4; ++j)
#pragma unroll
        for (int r = 0; r < 8; ++r)
          out[(size_t)(r0 + lrow[r]) * NN + lcol[j]] =
              ex[j][r] + TH * tanhf(acc[j][r] + bb[j]);
    }
    __syncthreads();
  }
}

// ---------------------------------------------------------------------------
extern "C" void kernel_launch(void* const* d_in, const int* in_sizes, int n_in,
                              void* d_out, int out_size, void* d_ws,
                              size_t ws_size, hipStream_t stream) {
  (void)in_sizes; (void)n_in; (void)out_size; (void)ws_size;
  const float* x    = (const float*)d_in[0];  // [256,512]
  const float* W    = (const float*)d_in[1];  // [512,512]
  const float* bias = (const float*)d_in[2];  // [512]
  float* out = (float*)d_out;
  _Float16* Apk = (_Float16*)d_ws;            // 512 KB packed A

  antisym_pack_kernel<<<(NN * NN) / 256, 256, 0, stream>>>(W, Apk);
  implicit_fused_kernel<<<16, 256, 0, stream>>>(x, Apk, bias, out);
}
